// GraphSAGE_5523327943254
// MI455X (gfx1250) — matrix operand order repaired and checked by reading, not compiled, not run on hardware
//
#include <hip/hip_runtime.h>
#include <hip/hip_bf16.h>

typedef __attribute__((ext_vector_type(16))) __bf16 v16bf;
typedef __attribute__((ext_vector_type(8)))  float  v8f;

#define N_NODES  100000
#define N_EDGES  3200000
#define IN_FEATS 768
#define HIDDEN   32
#define NCLS     8
#define KT1      (IN_FEATS / 32)   // 24 k-steps of 32
#define NT1      4                 // 64 outputs / 16
#define MTILES   (N_NODES / 16)    // 6250 (exact)

// ---------------------------------------------------------------------------
// Pack combined [W_neigh1; W_self1] (each [32,768] f32 row-major) into bf16
// WMMA B-fragments. Layout [kt][nt][lane][slot]:
//   N = nt*16 + (lane&15);  K = kt*32 + (slot&8)*2 + (lane>=16 ? 8 : 0) + (slot&7)
// ---------------------------------------------------------------------------
__global__ void pack_w1(const float* __restrict__ Wn, const float* __restrict__ Ws,
                        __bf16* __restrict__ wp) {
    int t = blockIdx.x * 256 + threadIdx.x;
    if (t >= KT1 * NT1 * 32 * 16) return;
    int s  = t & 15;
    int L  = (t >> 4) & 31;
    int nt = (t >> 9) & 3;
    int kt = t >> 11;
    int K = kt * 32 + (s & 8) * 2 + ((L >= 16) ? 8 : 0) + (s & 7);
    int n = nt * 16 + (L & 15);
    float v = (nt < 2) ? Wn[n * IN_FEATS + K] : Ws[(n - 32) * IN_FEATS + K];
    wp[t] = (__bf16)v;
}

// Combined [W_neigh2; W_self2] (each [8,32]) -> one 32x16 B-fragment.
__global__ void pack_w2(const float* __restrict__ Wn, const float* __restrict__ Ws,
                        __bf16* __restrict__ wp) {
    int t = blockIdx.x * 256 + threadIdx.x;
    if (t >= 512) return;
    int s = t & 15;
    int L = t >> 4;                       // 0..31
    int K = (s & 8) * 2 + ((L >= 16) ? 8 : 0) + (s & 7);
    int n = L & 15;
    float v = (n < 8) ? Wn[n * HIDDEN + K] : Ws[(n - 8) * HIDDEN + K];
    wp[t] = (__bf16)v;
}

// ---------------------------------------------------------------------------
// Fused layer-1 GEMM: H1pre[node][0..31] = x @ W_neigh1^T
//                     H1pre[node][32..63] = x @ W_self1^T
// One wave per 16-node tile. blockDim=64 (2 waves), grid=3125 -> 6250 waves.
// ---------------------------------------------------------------------------
__global__ __launch_bounds__(64) void gemm1(const float* __restrict__ x,
                                            const __bf16* __restrict__ wp,
                                            float* __restrict__ h1pre) {
    int wave = (blockIdx.x * 64 + threadIdx.x) >> 5;   // 0..6249
    int lane = threadIdx.x & 31;
    int lm   = lane & 15;
    int hi   = lane >> 4;                              // 0 or 1
    int row_base = wave * 16;
    const float* fp = x + (size_t)(row_base + lm) * IN_FEATS + hi * 8;

    v8f acc0 = {}, acc1 = {}, acc2 = {}, acc3 = {};

    for (int kt = 0; kt < KT1; ++kt) {
        // A fragment (16x32 bf16): slots 0-7 = K kb+koff.., slots 8-15 = K kb+16+koff..
        const float* p0 = fp + kt * 32;
        const float* p1 = p0 + 16;
        v16bf a;
#pragma unroll
        for (int i = 0; i < 8; ++i) {
            a[i]     = (__bf16)p0[i];
            a[8 + i] = (__bf16)p1[i];
        }
        const __bf16* wb = wp + ((size_t)(kt * 4) * 32 + lane) * 16;
        v16bf b0 = *(const v16bf*)(wb);
        v16bf b1 = *(const v16bf*)(wb + 512);
        v16bf b2 = *(const v16bf*)(wb + 1024);
        v16bf b3 = *(const v16bf*)(wb + 1536);
        acc0 = __builtin_amdgcn_wmma_f32_16x16x32_bf16(false, a, false, b0, (short)0, acc0, false, false);
        acc1 = __builtin_amdgcn_wmma_f32_16x16x32_bf16(false, a, false, b1, (short)0, acc1, false, false);
        acc2 = __builtin_amdgcn_wmma_f32_16x16x32_bf16(false, a, false, b2, (short)0, acc2, false, false);
        acc3 = __builtin_amdgcn_wmma_f32_16x16x32_bf16(false, a, false, b3, (short)0, acc3, false, false);
    }

    // D layout: lane -> N = lane&15; VGPR v -> M = v + 8*hi
#pragma unroll
    for (int v = 0; v < 8; ++v) {
        int node = row_base + v + 8 * hi;
        float* o = h1pre + (size_t)node * 64 + lm;
        o[0]  = acc0[v];
        o[16] = acc1[v];
        o[32] = acc2[v];
        o[48] = acc3[v];
    }
}

// ---------------------------------------------------------------------------
// Layer-1 edge scatter: (edge, c) threads, c in 0..31. Coalesced within edge.
// ---------------------------------------------------------------------------
__global__ void edge1(const int* __restrict__ src, const int* __restrict__ dst,
                      const float* __restrict__ h1pre,
                      float* __restrict__ agg1, float* __restrict__ deg) {
    long long t = (long long)blockIdx.x * 256 + threadIdx.x;
    if (t >= (long long)N_EDGES * 32) return;
    int c = (int)(t & 31);
    int e = (int)(t >> 5);
    int s = src[e], d = dst[e];
    atomicAdd(&agg1[(size_t)d * 32 + c], h1pre[(size_t)s * 64 + c]);
    if (c == 0) atomicAdd(&deg[d], 1.0f);
}

// ---------------------------------------------------------------------------
// Finalize layer 1 (mean-agg + self + bias, relu) and do layer-2 GEMM with a
// single WMMA per 16-node tile: h2pre[node][0..7]=neigh-pre, [8..15]=self.
// ---------------------------------------------------------------------------
__global__ __launch_bounds__(64) void fuse2(const float* __restrict__ h1pre,
                                            const float* __restrict__ agg1,
                                            const float* __restrict__ deg,
                                            const float* __restrict__ b1,
                                            const __bf16* __restrict__ wp2,
                                            float* __restrict__ h2pre) {
    int wave = (blockIdx.x * 64 + threadIdx.x) >> 5;   // 0..6249
    int lane = threadIdx.x & 31;
    int lm   = lane & 15;
    int hi   = lane >> 4;
    int row_base = wave * 16;
    int row = row_base + lm;
    float invd = 1.0f / fmaxf(deg[row], 1.0f);

    v16bf a;
    int c0 = hi * 8;
#pragma unroll
    for (int i = 0; i < 8; ++i) {
        int ca = c0 + i;        // slots 0..7
        int cb = 16 + c0 + i;   // slots 8..15
        float va = h1pre[(size_t)row * 64 + 32 + ca] + agg1[(size_t)row * 32 + ca] * invd + b1[ca];
        float vb = h1pre[(size_t)row * 64 + 32 + cb] + agg1[(size_t)row * 32 + cb] * invd + b1[cb];
        a[i]     = (__bf16)fmaxf(va, 0.0f);
        a[8 + i] = (__bf16)fmaxf(vb, 0.0f);
    }
    v16bf b = *(const v16bf*)(wp2 + (size_t)lane * 16);
    v8f acc = {};
    acc = __builtin_amdgcn_wmma_f32_16x16x32_bf16(false, a, false, b, (short)0, acc, false, false);
#pragma unroll
    for (int v = 0; v < 8; ++v) {
        int node = row_base + v + 8 * hi;
        h2pre[(size_t)node * 16 + lm] = acc[v];
    }
}

// Layer-2 edge scatter: (edge, c) threads, c in 0..7.
__global__ void edge2(const int* __restrict__ src, const int* __restrict__ dst,
                      const float* __restrict__ h2pre, float* __restrict__ agg2) {
    long long t = (long long)blockIdx.x * 256 + threadIdx.x;
    if (t >= (long long)N_EDGES * 8) return;
    int c = (int)(t & 7);
    int e = (int)(t >> 3);
    atomicAdd(&agg2[(size_t)dst[e] * 8 + c], h2pre[(size_t)src[e] * 16 + c]);
}

// Final: out = self + agg2/max(deg,1) + b2
__global__ void finalk(const float* __restrict__ h2pre, const float* __restrict__ agg2,
                       const float* __restrict__ deg, const float* __restrict__ b2,
                       float* __restrict__ out) {
    int t = blockIdx.x * 256 + threadIdx.x;
    if (t >= N_NODES * 8) return;
    int c = t & 7;
    int n = t >> 3;
    float invd = 1.0f / fmaxf(deg[n], 1.0f);
    out[t] = h2pre[(size_t)n * 16 + 8 + c] + agg2[t] * invd + b2[c];
}

extern "C" void kernel_launch(void* const* d_in, const int* in_sizes, int n_in,
                              void* d_out, int out_size, void* d_ws, size_t ws_size,
                              hipStream_t stream) {
    const float* feats = (const float*)d_in[0];
    const int*   src   = (const int*)d_in[1];
    const int*   dst   = (const int*)d_in[2];
    const float* Ws1   = (const float*)d_in[3];
    const float* Wn1   = (const float*)d_in[4];
    const float* b1    = (const float*)d_in[5];
    const float* Ws2   = (const float*)d_in[6];
    const float* Wn2   = (const float*)d_in[7];
    const float* b2    = (const float*)d_in[8];
    float* out = (float*)d_out;

    char* ws = (char*)d_ws;
    __bf16* wp1   = (__bf16*)(ws + 0);                 //  98,304 B
    __bf16* wp2   = (__bf16*)(ws + 98304);             //   1,024 B
    float*  h1pre = (float*)(ws + 99328);              // 25,600,000 B
    float*  agg1  = (float*)(ws + 25699328);           // 12,800,000 B
    float*  deg   = (float*)(ws + 38499328);           //    400,000 B
    float*  h2pre = (float*)(ws + 38899328);           //  6,400,000 B
    float*  agg2  = (float*)(ws + 45299328);           //  3,200,000 B  (total ~46.3 MB)

    // zero the accumulation buffers every call (graph replays accumulate otherwise)
    hipMemsetAsync(agg1, 0, (size_t)12800000 + 400000, stream);  // agg1 + deg contiguous
    hipMemsetAsync(agg2, 0, (size_t)3200000, stream);

    pack_w1<<<(KT1 * NT1 * 32 * 16 + 255) / 256, 256, 0, stream>>>(Wn1, Ws1, wp1);
    pack_w2<<<2, 256, 0, stream>>>(Wn2, Ws2, wp2);

    gemm1<<<MTILES / 2, 64, 0, stream>>>(feats, wp1, h1pre);

    edge1<<<(int)(((long long)N_EDGES * 32 + 255) / 256), 256, 0, stream>>>(src, dst, h1pre, agg1, deg);

    fuse2<<<MTILES / 2, 64, 0, stream>>>(h1pre, agg1, deg, b1, wp2, h2pre);

    edge2<<<(int)(((long long)N_EDGES * 8 + 255) / 256), 256, 0, stream>>>(src, dst, h2pre, agg2);

    finalk<<<(N_NODES * 8 + 255) / 256, 256, 0, stream>>>(h2pre, agg2, deg, b2, out);
}
